// MultiHeadSparseAttention_12034498363688
// MI455X (gfx1250) — matrix operand order repaired and checked by reading, not compiled, not run on hardware
//
#include <hip/hip_runtime.h>
#include <hip/hip_bf16.h>

typedef __bf16 bf16;
typedef __attribute__((ext_vector_type(16))) __bf16 v16bf;
typedef __attribute__((ext_vector_type(8)))  __bf16 v8bf;
typedef __attribute__((ext_vector_type(4)))  __bf16 v4bf;
typedef __attribute__((ext_vector_type(8)))  float  v8f;

#define WMMA_BF16(A_, B_, C_) \
  __builtin_amdgcn_wmma_f32_16x16x32_bf16(false, (A_), false, (B_), (short)0, (C_), false, false)

#define S_LEN   4096
#define DM      768
#define NHEAD   12
#define DEPTH   64
#define BANDH   128            // local_attn_ctx // 2
#define MTOT    8192           // B*S

// Concatenate two contiguous 16-byte halves into a 16-element bf16 fragment.
static __device__ __forceinline__ v16bf ldfrag(const bf16* p0, const bf16* p1) {
  v8bf a = *reinterpret_cast<const v8bf*>(p0);
  v8bf b = *reinterpret_cast<const v8bf*>(p1);
  return __builtin_shufflevector(a, b, 0, 1, 2, 3, 4, 5, 6, 7,
                                       8, 9, 10, 11, 12, 13, 14, 15);
}
// A-fragment (16x32, lane=M): two contiguous k-chunks at 8*hl and 16+8*hl.
static __device__ __forceinline__ v16bf ldfragA(const bf16* row, int hl) {
  return ldfrag(row + 8 * hl, row + 16 + 8 * hl);
}
// B-fragment (32x16, lane=N): two contiguous k-chunks at 8*hl and 16+8*hl.
static __device__ __forceinline__ v16bf ldfragB(const bf16* col, int hl) {
  return ldfrag(col + 8 * hl, col + 16 + 8 * hl);
}

// ---------------------------------------------------------------------------
// GEMM: Out(bf16) = X(f32,[8192,768]) @ W(f32,[768,768])
// VTRANS=0: head-major [b*H+h][s][64];  VTRANS=1: depth-major [b*H+h][dd][s]
// block 256 thr (8 waves); block tile 256(M) x 64(N); wave tile 32(M) x 64(N)
// ---------------------------------------------------------------------------
template <int VTRANS>
__global__ __launch_bounds__(256)
void proj_qkv_kernel(const float* __restrict__ X, const float* __restrict__ W,
                     bf16* __restrict__ Out) {
  __shared__ __align__(16) bf16 xs[256 * 32];   // [m][k] row-major  (16 KB)
  __shared__ __align__(16) bf16 wt[64 * 32];    // [n][k] transposed ( 4 KB)
  __shared__ __align__(16) bf16 sm[256 * 64];   // store bounce      (32 KB)
  const int m0 = blockIdx.x * 256;
  const int n0 = blockIdx.y * 64;
  const int t    = threadIdx.x;
  const int lane = t & 31;
  const int wave = t >> 5;
  const int hl   = lane >> 4;
  const int l15  = lane & 15;

  v8f acc[2][4];
  #pragma unroll
  for (int g = 0; g < 2; ++g)
    #pragma unroll
    for (int c = 0; c < 4; ++c)
      #pragma unroll
      for (int r = 0; r < 8; ++r) acc[g][c][r] = 0.0f;

  for (int k0 = 0; k0 < DM; k0 += 32) {
    if (k0 + 32 < DM)
      __builtin_prefetch(&X[(size_t)(m0 + wave * 32) * DM + k0 + 32], 0, 1);
    // X tile 256x32: float4 load -> packed bf16x4 LDS store
    for (int i = t * 4; i < 256 * 32; i += 256 * 4) {
      int r = i >> 5, c = i & 31;
      const float4 xv = *reinterpret_cast<const float4*>(
          &X[(size_t)(m0 + r) * DM + (k0 + c)]);
      v4bf pk;
      pk[0] = (bf16)xv.x; pk[1] = (bf16)xv.y;
      pk[2] = (bf16)xv.z; pk[3] = (bf16)xv.w;
      *reinterpret_cast<v4bf*>(&xs[i]) = pk;
    }
    // W tile 32x64 -> transposed LDS [n][k]
    for (int i = t; i < 32 * 64; i += 256) {
      int r = i >> 6, c = i & 63;
      wt[c * 32 + r] = (bf16)W[(size_t)(k0 + r) * DM + (n0 + c)];
    }
    __syncthreads();

    // batch all fragment loads, then 8 WMMAs
    const v16bf a0 = ldfragA(&xs[(wave * 32 + l15) * 32], hl);
    const v16bf a1 = ldfragA(&xs[(wave * 32 + 16 + l15) * 32], hl);
    const v16bf b0 = ldfragB(&wt[(0 * 16 + l15) * 32], hl);
    const v16bf b1 = ldfragB(&wt[(1 * 16 + l15) * 32], hl);
    const v16bf b2 = ldfragB(&wt[(2 * 16 + l15) * 32], hl);
    const v16bf b3 = ldfragB(&wt[(3 * 16 + l15) * 32], hl);
    acc[0][0] = WMMA_BF16(a0, b0, acc[0][0]);
    acc[0][1] = WMMA_BF16(a0, b1, acc[0][1]);
    acc[0][2] = WMMA_BF16(a0, b2, acc[0][2]);
    acc[0][3] = WMMA_BF16(a0, b3, acc[0][3]);
    acc[1][0] = WMMA_BF16(a1, b0, acc[1][0]);
    acc[1][1] = WMMA_BF16(a1, b1, acc[1][1]);
    acc[1][2] = WMMA_BF16(a1, b2, acc[1][2]);
    acc[1][3] = WMMA_BF16(a1, b3, acc[1][3]);
    __syncthreads();
  }

  // ---- epilogue: stage to LDS in output-friendly layout, store b128 ----
  #pragma unroll
  for (int g = 0; g < 2; ++g)
    #pragma unroll
    for (int c = 0; c < 4; ++c)
      #pragma unroll
      for (int r = 0; r < 8; ++r) {
        const int m  = wave * 32 + g * 16 + r + 8 * hl;   // 0..255
        const int nn = c * 16 + l15;                      // 0..63
        if (VTRANS) sm[nn * 256 + m] = (bf16)acc[g][c][r];
        else        sm[m * 64 + nn]  = (bf16)acc[g][c][r];
      }
  __syncthreads();

  const int bb = m0 >> 12;          // batch (4096 rows each, 256 | 4096)
  const int h  = n0 >> 6;           // whole block sits in one head
  const int s0 = m0 & (S_LEN - 1);
  if (VTRANS) {
    bf16* obase = Out + ((size_t)(bb * NHEAD + h) * DEPTH) * S_LEN + s0;
    for (int i = t * 8; i < 256 * 64; i += 256 * 8) {
      const int dd = i >> 8, ssl = i & 255;
      *reinterpret_cast<v8bf*>(&obase[(size_t)dd * S_LEN + ssl]) =
          *reinterpret_cast<const v8bf*>(&sm[i]);
    }
  } else {
    bf16* obase = Out + ((size_t)(bb * NHEAD + h) * S_LEN + s0) * DEPTH;
    for (int i = t * 8; i < 256 * 64; i += 256 * 8) {
      *reinterpret_cast<v8bf*>(&obase[i]) =
          *reinterpret_cast<const v8bf*>(&sm[i]);
    }
  }
}

// ---------------------------------------------------------------------------
// GEMM: Out(f32,[8192,768]) = A(bf16,[8192,768]) @ W(f32,[768,768])
// ---------------------------------------------------------------------------
__global__ __launch_bounds__(256)
void out_proj_kernel(const bf16* __restrict__ Ain, const float* __restrict__ W,
                     float* __restrict__ Out) {
  __shared__ __align__(16) bf16  xs[256 * 32];
  __shared__ __align__(16) bf16  wt[64 * 32];
  __shared__ __align__(16) float smf[256 * 64];  // 64 KB f32 bounce
  const int m0 = blockIdx.x * 256;
  const int n0 = blockIdx.y * 64;
  const int t    = threadIdx.x;
  const int lane = t & 31;
  const int wave = t >> 5;
  const int hl   = lane >> 4;
  const int l15  = lane & 15;

  v8f acc[2][4];
  #pragma unroll
  for (int g = 0; g < 2; ++g)
    #pragma unroll
    for (int c = 0; c < 4; ++c)
      #pragma unroll
      for (int r = 0; r < 8; ++r) acc[g][c][r] = 0.0f;

  for (int k0 = 0; k0 < DM; k0 += 32) {
    if (k0 + 32 < DM)
      __builtin_prefetch(&Ain[(size_t)(m0 + wave * 32) * DM + k0 + 32], 0, 1);
    for (int i = t * 8; i < 256 * 32; i += 256 * 8) {
      int r = i >> 5, c = i & 31;
      *reinterpret_cast<v8bf*>(&xs[i]) = *reinterpret_cast<const v8bf*>(
          &Ain[(size_t)(m0 + r) * DM + (k0 + c)]);
    }
    for (int i = t; i < 32 * 64; i += 256) {
      int r = i >> 6, c = i & 63;
      wt[c * 32 + r] = (bf16)W[(size_t)(k0 + r) * DM + (n0 + c)];
    }
    __syncthreads();

    const v16bf a0 = ldfragA(&xs[(wave * 32 + l15) * 32], hl);
    const v16bf a1 = ldfragA(&xs[(wave * 32 + 16 + l15) * 32], hl);
    const v16bf b0 = ldfragB(&wt[(0 * 16 + l15) * 32], hl);
    const v16bf b1 = ldfragB(&wt[(1 * 16 + l15) * 32], hl);
    const v16bf b2 = ldfragB(&wt[(2 * 16 + l15) * 32], hl);
    const v16bf b3 = ldfragB(&wt[(3 * 16 + l15) * 32], hl);
    acc[0][0] = WMMA_BF16(a0, b0, acc[0][0]);
    acc[0][1] = WMMA_BF16(a0, b1, acc[0][1]);
    acc[0][2] = WMMA_BF16(a0, b2, acc[0][2]);
    acc[0][3] = WMMA_BF16(a0, b3, acc[0][3]);
    acc[1][0] = WMMA_BF16(a1, b0, acc[1][0]);
    acc[1][1] = WMMA_BF16(a1, b1, acc[1][1]);
    acc[1][2] = WMMA_BF16(a1, b2, acc[1][2]);
    acc[1][3] = WMMA_BF16(a1, b3, acc[1][3]);
    __syncthreads();
  }

  // ---- epilogue: f32 LDS bounce -> coalesced float4 stores ----
  #pragma unroll
  for (int g = 0; g < 2; ++g)
    #pragma unroll
    for (int c = 0; c < 4; ++c)
      #pragma unroll
      for (int r = 0; r < 8; ++r) {
        const int m  = wave * 32 + g * 16 + r + 8 * hl;
        const int nn = c * 16 + l15;
        smf[m * 64 + nn] = acc[g][c][r];
      }
  __syncthreads();

  for (int i = t * 4; i < 256 * 64; i += 256 * 4) {
    const int rr = i >> 6, cc = i & 63;
    *reinterpret_cast<float4*>(&Out[(size_t)(m0 + rr) * DM + n0 + cc]) =
        *reinterpret_cast<const float4*>(&smf[i]);
  }
}

// ---------------------------------------------------------------------------
// Banded attention. One wave per (bh, 16-query tile). Online softmax over
// 32-key tiles. Q/K head-major [bh][s][64]; V depth-major [bh][dd][s];
// out merged-head bf16 [b*S+s][768]. All WMMA fragments are b128-class loads.
// ---------------------------------------------------------------------------
__global__ __launch_bounds__(32)
void attn_kernel(const bf16* __restrict__ Q, const bf16* __restrict__ K,
                 const bf16* __restrict__ Vt, bf16* __restrict__ Aout) {
  __shared__ __align__(16) bf16 p_lds[16 * 32];
  const int q0  = blockIdx.x * 16;
  const int bh  = blockIdx.y;
  const int lane = threadIdx.x;
  const int hl   = lane >> 4;
  const int l15  = lane & 15;
  const size_t base = (size_t)bh * S_LEN * DEPTH;
  const float scale = 0.125f;   // 1/sqrt(64)

  v16bf qa[2];
  #pragma unroll
  for (int kc = 0; kc < 2; ++kc)
    qa[kc] = ldfragA(&Q[base + (size_t)(q0 + l15) * DEPTH + kc * 32], hl);

  v8f o[4];
  float mrow[8], lrow[8];
  #pragma unroll
  for (int c = 0; c < 4; ++c)
    #pragma unroll
    for (int r = 0; r < 8; ++r) o[c][r] = 0.0f;
  #pragma unroll
  for (int r = 0; r < 8; ++r) { mrow[r] = -1e30f; lrow[r] = 0.0f; }

  const int ks = (q0 >= BANDH) ? ((q0 - BANDH) & ~31) : 0;
  int ke = q0 + 16 + BANDH;
  if (ke > S_LEN) ke = S_LEN;

  for (int kt = ks; kt < ke; kt += 32) {
    v8f sc[2];
    #pragma unroll
    for (int sub = 0; sub < 2; ++sub) {
      #pragma unroll
      for (int r = 0; r < 8; ++r) sc[sub][r] = 0.0f;
      const bf16* krow = &K[base + (size_t)(kt + sub * 16 + l15) * DEPTH];
      #pragma unroll
      for (int kc = 0; kc < 2; ++kc) {
        const v16bf kb = ldfragB(krow + kc * 32, hl);
        sc[sub] = WMMA_BF16(qa[kc], kb, sc[sub]);
      }
    }

    const int j0 = kt + l15;
    const int j1 = kt + 16 + l15;
    #pragma unroll
    for (int r = 0; r < 8; ++r) {
      const int i = q0 + r + 8 * hl;
      int d0 = i - j0; if (d0 < 0) d0 = -d0;
      int d1 = i - j1; if (d1 < 0) d1 = -d1;
      const bool ok0 = (d0 <= BANDH);
      const bool ok1 = (d1 <= BANDH);
      float v0 = ok0 ? sc[0][r] * scale : -1e30f;
      float v1 = ok1 ? sc[1][r] * scale : -1e30f;
      float tm = fmaxf(v0, v1);
      #pragma unroll
      for (int w = 1; w < 16; w <<= 1) tm = fmaxf(tm, __shfl_xor(tm, w, 32));
      const float mnew = fmaxf(mrow[r], tm);
      const float p0 = ok0 ? __expf(v0 - mnew) : 0.0f;
      const float p1 = ok1 ? __expf(v1 - mnew) : 0.0f;
      float ts = p0 + p1;
      #pragma unroll
      for (int w = 1; w < 16; w <<= 1) ts += __shfl_xor(ts, w, 32);
      const float alpha = __expf(mrow[r] - mnew);
      lrow[r] = lrow[r] * alpha + ts;
      mrow[r] = mnew;
      #pragma unroll
      for (int c = 0; c < 4; ++c) o[c][r] *= alpha;
      sc[0][r] = p0;
      sc[1][r] = p1;
    }

    #pragma unroll
    for (int sub = 0; sub < 2; ++sub)
      #pragma unroll
      for (int r = 0; r < 8; ++r)
        p_lds[(r + 8 * hl) * 32 + sub * 16 + l15] = (bf16)sc[sub][r];
    __syncthreads();

    const v16bf pa = ldfragA(&p_lds[l15 * 32], hl);
    #pragma unroll
    for (int c = 0; c < 4; ++c) {
      const v16bf vb =
          ldfragB(&Vt[base + (size_t)(c * 16 + l15) * S_LEN + kt], hl);
      o[c] = WMMA_BF16(pa, vb, o[c]);
    }
    __syncthreads();
  }

  const int bb = bh / NHEAD, h = bh % NHEAD;
  #pragma unroll
  for (int c = 0; c < 4; ++c) {
    #pragma unroll
    for (int r = 0; r < 8; ++r) {
      const int srow = q0 + r + 8 * hl;
      const float ov = o[c][r] / fmaxf(lrow[r], 1e-30f);
      size_t idx = ((size_t)(bb * S_LEN + srow)) * DM + h * DEPTH + c * 16 + l15;
      Aout[idx] = (bf16)ov;
    }
  }
}

// ---------------------------------------------------------------------------
extern "C" void kernel_launch(void* const* d_in, const int* in_sizes, int n_in,
                              void* d_out, int out_size, void* d_ws, size_t ws_size,
                              hipStream_t stream) {
  (void)in_sizes; (void)n_in; (void)out_size; (void)ws_size;
  const float* x  = (const float*)d_in[0];
  const float* Wq = (const float*)d_in[1];
  const float* Wk = (const float*)d_in[2];
  const float* Wv = (const float*)d_in[3];
  const float* Wo = (const float*)d_in[4];
  float* out = (float*)d_out;

  const size_t mat = (size_t)MTOT * DM;   // 8192*768 elements
  bf16* Qb = (bf16*)d_ws;
  bf16* Kb = Qb + mat;
  bf16* Vb = Kb + mat;                    // depth-major
  bf16* Ab = Vb + mat;

  dim3 gproj(MTOT / 256, DM / 64), bproj(256);
  proj_qkv_kernel<0><<<gproj, bproj, 0, stream>>>(x, Wq, Qb);
  proj_qkv_kernel<0><<<gproj, bproj, 0, stream>>>(x, Wk, Kb);
  proj_qkv_kernel<1><<<gproj, bproj, 0, stream>>>(x, Wv, Vb);

  attn_kernel<<<dim3(S_LEN / 16, 2 * NHEAD), 32, 0, stream>>>(Qb, Kb, Vb, Ab);

  out_proj_kernel<<<gproj, bproj, 0, stream>>>(Ab, Wo, out);
}
